// FastAttention_10780367913085
// MI455X (gfx1250) — matrix operand order repaired and checked by reading, not compiled, not run on hardware
//
#include <hip/hip_runtime.h>
#include <hip/hip_bf16.h>

// ---------------------------------------------------------------------------
// FastAttention for MI455X (gfx1250, wave32).
// Bandwidth-bound (~71 GFLOP vs ~250MB traffic @ 23.3 TB/s): f16 staging in a
// ~131MB workspace (L2-resident), all GEMMs via v_wmma_f32_16x16x32_f16 with
// operand tiles DMA'd into LDS by the Tensor Data Mover (tensor_load_to_lds,
// double-buffered, s_wait_tensorcnt), bias/residual fused into WMMA stores.
// Wave tile 32x64 (8 accumulators) so LDS latency hides under the WMMA chain.
// ---------------------------------------------------------------------------

typedef __attribute__((ext_vector_type(16))) _Float16 v16h;
typedef __attribute__((ext_vector_type(8)))  _Float16 v8h;
typedef __attribute__((ext_vector_type(8)))  float    v8f;
typedef __attribute__((ext_vector_type(4)))  int      v4i;
typedef __attribute__((ext_vector_type(8)))  int      v8i;

#define B_DIM      4
#define N_SEQ      8192
#define DIM        512
#define HEADS      8
#define DIM_HEAD   64
#define INNER      512
#define QKV_N      1536
#define M_ROWS     (B_DIM * N_SEQ)          // 32768
#define HEAD_ELEMS (N_SEQ * DIM_HEAD)       // 524288 per (b,h)
#define BH         (B_DIM * HEADS)          // 32
#define SCALE_F    0.125f                   // 64^-0.5
#define KC         64                       // K-chunk staged per TDM transfer
#define MT         128                      // block tile rows (4 waves x 32)
#define NT         64                       // block tile cols

// ---- workspace layout (bytes, all 256-aligned) ----------------------------
#define SZ_H(n)    ((size_t)(n) * 2)
#define OFF_XH     ((size_t)0)
#define OFF_QH     (OFF_XH    + SZ_H(M_ROWS * 512))
#define OFF_KH     (OFF_QH    + SZ_H(M_ROWS * 512))
#define OFF_VH     (OFF_KH    + SZ_H(M_ROWS * 512))
#define OFF_WQKVH  (OFF_VH    + SZ_H(M_ROWS * 512))
#define OFF_WRH    (OFF_WQKVH + SZ_H(QKV_N * 512))
#define OFF_WOUTH  (OFF_WRH   + SZ_H(64 * 64))
#define OFF_GQ     (OFF_WOUTH + SZ_H(512 * 512))
#define OFF_GK     (OFF_GQ    + (size_t)BH * 64 * 4)
#define OFF_LOGITS (OFF_GK    + (size_t)BH * 64 * 4)
#define OFF_VGH    OFF_XH                    // alias: x dead after qkv GEMMs
#define OFF_RH     OFF_KH                    // alias: k dead after stage2

// ---------------------------------------------------------------------------
// Tensor Data Mover: 2D tile (rows x tile_k f16, row stride = stride_elems)
// from global memory into LDS at byte offset lds_byte_off. D# per ISA 8.3/8.4.
// ---------------------------------------------------------------------------
__device__ __forceinline__ void tdm_load_2d(unsigned int lds_byte_off,
                                            const _Float16* gptr,
                                            unsigned int tile_k,
                                            unsigned int rows,
                                            unsigned int stride_elems) {
  unsigned long long ga = (unsigned long long)(uintptr_t)gptr;
  v4i g0;
  g0[0] = 1;                                        // count=1, user descriptor
  g0[1] = (int)lds_byte_off;                        // lds_addr
  g0[2] = (int)(unsigned int)ga;                    // global_addr[31:0]
  g0[3] = (int)((((unsigned int)(ga >> 32)) & 0x01FFFFFFu) | 0x80000000u); // +type=2
  v8i g1;
  g1[0] = 0x00010000;                               // data_size = 2 bytes
  g1[1] = (int)((tile_k & 0xFFFFu) << 16);          // tensor_dim0[15:0]
  g1[2] = (int)(((tile_k >> 16) & 0xFFFFu) | ((rows & 0xFFFFu) << 16));
  g1[3] = (int)(((rows >> 16) & 0xFFFFu) | ((tile_k & 0xFFFFu) << 16)); // tile_dim0
  g1[4] = (int)(rows & 0xFFFFu);                    // tile_dim1 (tile_dim2=0)
  g1[5] = (int)stride_elems;                        // tensor_dim0_stride[31:0]
  g1[6] = 0;
  g1[7] = 0;
  asm volatile("tensor_load_to_lds %0, %1" :: "s"(g0), "s"(g1) : "memory");
}

__device__ __forceinline__ unsigned int lds_off_u32(const void* p) {
  // addrspace(3) offsets live in the low 32 bits of the generic address
  return (unsigned int)(uintptr_t)p;
}

// ---------------------------------------------------------------------------
// WMMA fragment helpers (A and B both use contiguous k = 16*(lane>>4) + e), operands read from LDS.
// ---------------------------------------------------------------------------
__device__ __forceinline__ v16h load_frag_a_lds(const _Float16* s, int mrow0,
                                                int kk, int lane) {
  int m  = mrow0 + (lane & 15);
  int kb = kk + ((lane >> 4) << 4);                 // 0 or 16
  const _Float16* p = s + m * KC + kb;
  v8h lo = *(const v8h*)(p);                        // K = kb .. kb+7
  v8h hi = *(const v8h*)(p + 8);                    // K = kb+8 .. kb+15
  v16h r;
#pragma unroll
  for (int e = 0; e < 8; ++e) { r[e] = lo[e]; r[8 + e] = hi[e]; }
  return r;
}

// B fragment for C = A * W^T: lane holds column n = lane&15, contiguous K
// range of 16 starting at (lane>>4)*16.
__device__ __forceinline__ v16h load_frag_b_lds(const _Float16* s, int n0,
                                                int kk, int lane) {
  int n  = n0 + (lane & 15);
  int k2 = kk + ((lane >> 4) << 4);                 // 0 or 16
  return *(const v16h*)(s + n * KC + k2);
}

__device__ __forceinline__ v8f wmma_f16(v16h a, v16h b, v8f c) {
  return __builtin_amdgcn_wmma_f32_16x16x32_f16(
      /*neg_a=*/false, a, /*neg_b=*/false, b,
      /*c_mod=*/(short)0, c, /*reuse_a=*/false, /*reuse_b=*/false);
}

template <typename OutT>
__device__ __forceinline__ void store_tile(OutT* __restrict__ C, int ldc,
                                           int m0, int n0, int lane, v8f c,
                                           const float* __restrict__ bias,
                                           const _Float16* __restrict__ resid) {
  int n     = n0 + (lane & 15);
  int mbase = m0 + ((lane >> 4) << 3);
  float bv = bias ? bias[n] : 0.0f;
#pragma unroll
  for (int i = 0; i < 8; ++i) {
    size_t idx = (size_t)(mbase + i) * ldc + n;
    float val = c[i] + bv;
    if (resid) val += (float)resid[idx];
    C[idx] = (OutT)val;
  }
}

// ---------------------------------------------------------------------------
// Kernel: f32 -> f16 conversion
// ---------------------------------------------------------------------------
__global__ void fa_cvt_f16(const float* __restrict__ src,
                           _Float16* __restrict__ dst, int n) {
  int i = blockIdx.x * blockDim.x + threadIdx.x;
  if (i < n) dst[i] = (_Float16)src[i];
}

// ---------------------------------------------------------------------------
// Kernel: GEMM  C = A * W^T (+bias) (+resid); A:(M,K) f16, W:(N,K) f16.
// Block = 128 thr (4 waves), block tile 128x64, wave tile 32x64 (8 accs).
// A(128xKC) and B(64xKC) chunks double-buffered in LDS via TDM: wave 0
// issues tensor_load_to_lds for chunk c+1, waits TENSORcnt for chunk c,
// workgroup barrier publishes; WMMA consumes via ds_load_b128.
// ---------------------------------------------------------------------------
template <typename OutT>
__global__ void __launch_bounds__(128)
fa_gemm_awt(const _Float16* __restrict__ A, int lda,
            const _Float16* __restrict__ W, int ldw,
            OutT* __restrict__ C, int ldc,
            const float* __restrict__ bias,
            const _Float16* __restrict__ resid, int K) {
  __shared__ __attribute__((aligned(32))) _Float16 sA[2][MT * KC];  // 2x16KB
  __shared__ __attribute__((aligned(32))) _Float16 sB[2][NT * KC];  // 2x 8KB
  int lane = threadIdx.x & 31;
  int wave = threadIdx.x >> 5;
  int m0 = blockIdx.x * MT;
  int n0 = blockIdx.y * NT;
  int nchunks = K / KC;
  bool issuer = (wave == 0);

  if (issuer) {
    tdm_load_2d(lds_off_u32(sA[0]), A + (size_t)m0 * lda, KC, MT, lda);
    tdm_load_2d(lds_off_u32(sB[0]), W + (size_t)n0 * ldw, KC, NT, ldw);
  }

  v8f a00 = {}, a01 = {}, a02 = {}, a03 = {};   // rows wave*32 .. +15
  v8f a10 = {}, a11 = {}, a12 = {}, a13 = {};   // rows wave*32+16 .. +31
  for (int c = 0; c < nchunks; ++c) {
    if (issuer) __builtin_amdgcn_s_wait_tensorcnt(0);   // chunk c landed
    __syncthreads();                                    // publish LDS to block
    if (issuer && (c + 1 < nchunks)) {                  // prefetch chunk c+1
      int k0 = (c + 1) * KC;
      tdm_load_2d(lds_off_u32(sA[(c + 1) & 1]), A + (size_t)m0 * lda + k0,
                  KC, MT, lda);
      tdm_load_2d(lds_off_u32(sB[(c + 1) & 1]), W + (size_t)n0 * ldw + k0,
                  KC, NT, ldw);
    }
    const _Float16* a_s = sA[c & 1];
    const _Float16* b_s = sB[c & 1];
#pragma unroll
    for (int kk = 0; kk < KC; kk += 32) {
      // 6 fragment loads feed 8 WMMAs; next kk's loads overlap this chain.
      v16h fa0 = load_frag_a_lds(a_s, wave * 32,      kk, lane);
      v16h fa1 = load_frag_a_lds(a_s, wave * 32 + 16, kk, lane);
      v16h fb0 = load_frag_b_lds(b_s, 0,  kk, lane);
      v16h fb1 = load_frag_b_lds(b_s, 16, kk, lane);
      v16h fb2 = load_frag_b_lds(b_s, 32, kk, lane);
      v16h fb3 = load_frag_b_lds(b_s, 48, kk, lane);
      a00 = wmma_f16(fa0, fb0, a00);
      a01 = wmma_f16(fa0, fb1, a01);
      a02 = wmma_f16(fa0, fb2, a02);
      a03 = wmma_f16(fa0, fb3, a03);
      a10 = wmma_f16(fa1, fb0, a10);
      a11 = wmma_f16(fa1, fb1, a11);
      a12 = wmma_f16(fa1, fb2, a12);
      a13 = wmma_f16(fa1, fb3, a13);
    }
    __syncthreads();                 // compute done before chunk c+2 overwrite
  }

  int mw = m0 + wave * 32;
  store_tile<OutT>(C, ldc, mw,      n0,      lane, a00, bias, resid);
  store_tile<OutT>(C, ldc, mw,      n0 + 16, lane, a01, bias, resid);
  store_tile<OutT>(C, ldc, mw,      n0 + 32, lane, a02, bias, resid);
  store_tile<OutT>(C, ldc, mw,      n0 + 48, lane, a03, bias, resid);
  store_tile<OutT>(C, ldc, mw + 16, n0,      lane, a10, bias, resid);
  store_tile<OutT>(C, ldc, mw + 16, n0 + 16, lane, a11, bias, resid);
  store_tile<OutT>(C, ldc, mw + 16, n0 + 32, lane, a12, bias, resid);
  store_tile<OutT>(C, ldc, mw + 16, n0 + 48, lane, a13, bias, resid);
}

// ---------------------------------------------------------------------------
// Kernel: per-(b,h) softmax-pooled global vectors (two-pass, block=256).
// Head data is a contiguous (8192,64) f16 block thanks to the raw reshape.
// ---------------------------------------------------------------------------
__device__ void fa_softmax_wmean(const _Float16* __restrict__ X,
                                 const unsigned char* __restrict__ mrow,
                                 float* __restrict__ logits,
                                 const float* __restrict__ wv,
                                 float* __restrict__ outv,
                                 float* __restrict__ red,
                                 float* __restrict__ accs,
                                 float* __restrict__ psums, int tid) {
  const float NEG = -3.0e38f;
  float lmax = NEG;
  for (int n = tid; n < N_SEQ; n += 256) {
    const v8h* r8 = (const v8h*)(X + (size_t)n * 64);
    float l = 0.f;
#pragma unroll
    for (int c = 0; c < 8; ++c) {
      v8h h = r8[c];
#pragma unroll
      for (int e = 0; e < 8; ++e) l += (float)h[e] * wv[c * 8 + e];
    }
    l *= SCALE_F;
    if (mrow[n]) l = NEG;
    logits[n] = l;
    lmax = fmaxf(lmax, l);
  }
  red[tid] = lmax;
  __syncthreads();
  for (int s = 128; s > 0; s >>= 1) {
    if (tid < s) red[tid] = fmaxf(red[tid], red[tid + s]);
    __syncthreads();
  }
  float m = red[0];
  __syncthreads();
  int d = tid & 63, g = tid >> 6;
  float acc = 0.f, ps = 0.f;
  for (int n = g; n < N_SEQ; n += 4) {
    float p = __expf(logits[n] - m);
    acc += p * (float)X[(size_t)n * 64 + d];
    ps += p;
  }
  accs[tid] = acc;
  if (d == 0) psums[g] = ps;
  __syncthreads();
  if (tid < 64) {
    float a = accs[tid] + accs[64 + tid] + accs[128 + tid] + accs[192 + tid];
    float P = psums[0] + psums[1] + psums[2] + psums[3];
    outv[tid] = a / P;
  }
  __syncthreads();
}

__global__ void __launch_bounds__(256)
fa_stage2(const _Float16* __restrict__ qh, const _Float16* __restrict__ kh,
          const float* __restrict__ wq, const float* __restrict__ wk,
          const unsigned char* __restrict__ mask,
          float* __restrict__ gq, float* __restrict__ gk,
          float* __restrict__ logits_ws) {
  __shared__ float s_red[256];
  __shared__ float s_acc[256];
  __shared__ float s_psum[4];
  __shared__ float s_w[64];
  __shared__ float s_out[64];
  int bh  = blockIdx.x;
  int b   = bh >> 3;
  int tid = threadIdx.x;
  const _Float16* Q = qh + (size_t)bh * HEAD_ELEMS;
  const _Float16* K = kh + (size_t)bh * HEAD_ELEMS;
  const unsigned char* mrow = mask + (size_t)b * N_SEQ;
  float* logits = logits_ws + (size_t)bh * N_SEQ;

  if (tid < 64) s_w[tid] = wq[tid];
  __syncthreads();
  fa_softmax_wmean(Q, mrow, logits, s_w, s_out, s_red, s_acc, s_psum, tid);
  if (tid < 64) gq[bh * 64 + tid] = s_out[tid];
  __syncthreads();
  if (tid < 64) s_w[tid] = s_out[tid] * wk[tid];     // global_q[d] * w_k[d]
  __syncthreads();
  fa_softmax_wmean(K, mrow, logits, s_w, s_out, s_red, s_acc, s_psum, tid);
  if (tid < 64) gk[bh * 64 + tid] = s_out[tid];
}

// ---------------------------------------------------------------------------
// Kernel: vg = v * global_k (broadcast over the head's 64-dim)
// ---------------------------------------------------------------------------
__global__ void fa_vg(const _Float16* __restrict__ vh,
                      const float* __restrict__ gk,
                      _Float16* __restrict__ vgh, int total) {
  int i = blockIdx.x * blockDim.x + threadIdx.x;
  if (i >= total) return;
  int bh = i >> 19;                 // HEAD_ELEMS = 2^19
  int d  = i & 63;
  vgh[i] = (_Float16)((float)vh[i] * gk[bh * 64 + d]);
}

// ---------------------------------------------------------------------------
extern "C" void kernel_launch(void* const* d_in, const int* in_sizes, int n_in,
                              void* d_out, int out_size, void* d_ws,
                              size_t ws_size, hipStream_t stream) {
  const float*         x     = (const float*)d_in[0];
  const unsigned char* mask  = (const unsigned char*)d_in[1];
  const float*         W_qkv = (const float*)d_in[2];
  const float*         wq    = (const float*)d_in[3];
  const float*         wk    = (const float*)d_in[4];
  const float*         W_r   = (const float*)d_in[5];
  const float*         b_r   = (const float*)d_in[6];
  const float*         W_out = (const float*)d_in[7];
  const float*         b_out = (const float*)d_in[8];
  float* out = (float*)d_out;

  char* ws = (char*)d_ws;
  _Float16* xh     = (_Float16*)(ws + OFF_XH);
  _Float16* qh     = (_Float16*)(ws + OFF_QH);
  _Float16* kh     = (_Float16*)(ws + OFF_KH);
  _Float16* vh     = (_Float16*)(ws + OFF_VH);
  _Float16* wqkvh  = (_Float16*)(ws + OFF_WQKVH);
  _Float16* wrh    = (_Float16*)(ws + OFF_WRH);
  _Float16* wouth  = (_Float16*)(ws + OFF_WOUTH);
  float*    gq     = (float*)(ws + OFF_GQ);
  float*    gk     = (float*)(ws + OFF_GK);
  float*    logits = (float*)(ws + OFF_LOGITS);
  _Float16* vgh    = (_Float16*)(ws + OFF_VGH);   // aliases xh (dead)
  _Float16* rh     = (_Float16*)(ws + OFF_RH);    // aliases kh (dead)

  // 1) f32 -> f16 staging
  {
    int n = M_ROWS * DIM;
    fa_cvt_f16<<<(n + 255) / 256, 256, 0, stream>>>(x, xh, n);
  }
  {
    int n = QKV_N * DIM;
    fa_cvt_f16<<<(n + 255) / 256, 256, 0, stream>>>(W_qkv, wqkvh, n);
  }
  {
    int n = DIM_HEAD * DIM_HEAD;
    fa_cvt_f16<<<(n + 255) / 256, 256, 0, stream>>>(W_r, wrh, n);
  }
  {
    int n = DIM * INNER;
    fa_cvt_f16<<<(n + 255) / 256, 256, 0, stream>>>(W_out, wouth, n);
  }

  // 2) q/k/v = x @ W_qkv[part]^T   (three TDM-staged WMMA GEMMs, N=512 each)
  fa_gemm_awt<_Float16><<<dim3(M_ROWS / MT, DIM / NT), 128, 0, stream>>>(
      xh, DIM, wqkvh + (size_t)0 * 512 * 512, DIM, qh, 512,
      (const float*)nullptr, (const _Float16*)nullptr, DIM);
  fa_gemm_awt<_Float16><<<dim3(M_ROWS / MT, DIM / NT), 128, 0, stream>>>(
      xh, DIM, wqkvh + (size_t)1 * 512 * 512, DIM, kh, 512,
      (const float*)nullptr, (const _Float16*)nullptr, DIM);
  fa_gemm_awt<_Float16><<<dim3(M_ROWS / MT, DIM / NT), 128, 0, stream>>>(
      xh, DIM, wqkvh + (size_t)2 * 512 * 512, DIM, vh, 512,
      (const float*)nullptr, (const _Float16*)nullptr, DIM);

  // 3) softmax-pooled global_q / global_k per (b,h)
  fa_stage2<<<BH, 256, 0, stream>>>(qh, kh, wq, wk, mask, gq, gk, logits);

  // 4) vg = v * global_k
  {
    int n = M_ROWS * DIM;
    fa_vg<<<(n + 255) / 256, 256, 0, stream>>>(vh, gk, vgh, n);
  }

  // 5) r = vg @ W_r^T + b_r + q   (M=262144, N=64, K=64; residual fused)
  fa_gemm_awt<_Float16><<<dim3((BH * N_SEQ) / MT, 1), 128, 0, stream>>>(
      vgh, 64, wrh, 64, rh, 64, b_r, qh, 64);

  // 6) out = r @ W_out^T + b_out   (M=32768, N=512, K=512; f32 out)
  fa_gemm_awt<float><<<dim3(M_ROWS / MT, DIM / NT), 128, 0, stream>>>(
      rh, 512, wouth, 512, out, 512, b_out, (const _Float16*)nullptr, 512);
}
